// HTGNNLayer_42863773614193
// MI455X (gfx1250) — matrix operand order repaired and checked by reading, not compiled, not run on hardware
//
#include <hip/hip_runtime.h>
#include <hip/hip_bf16.h>

#define T_STEPS 8
#define NNODES 100000
#define NH 128
#define NI 64
#define NODES_PER_WG 16
#define THREADS 256
#define QLD 136   // padded LDS row stride (halves) for Q/K/V

typedef __attribute__((ext_vector_type(16))) __bf16 v16bf;
typedef __attribute__((ext_vector_type(8)))  float  v8f;
typedef __attribute__((ext_vector_type(4)))  unsigned uint4v;

__device__ __forceinline__ unsigned short f2bf_us(float f) {
  unsigned u = __builtin_bit_cast(unsigned, f);
  unsigned r = u + 0x7FFFu + ((u >> 16) & 1u);   // round-to-nearest-even
  return (unsigned short)(r >> 16);
}
__device__ __forceinline__ float bf2f(unsigned short h) {
  return __builtin_bit_cast(float, ((unsigned)h) << 16);
}
__device__ __forceinline__ uint2 pack4bf(float4 v) {
  uint2 r;
  r.x = (unsigned)f2bf_us(v.x) | ((unsigned)f2bf_us(v.y) << 16);
  r.y = (unsigned)f2bf_us(v.z) | ((unsigned)f2bf_us(v.w) << 16);
  return r;
}

// fp32 weight -> bf16 LDS, vectorized (n multiple of 4)
__device__ __forceinline__ void load_w_bf16(unsigned short* dst, const float* src,
                                            int n, int tid) {
  const float4* s4 = reinterpret_cast<const float4*>(src);
  uint2* d2 = reinterpret_cast<uint2*>(dst);
  for (int i = tid; i < (n >> 2); i += THREADS) d2[i] = pack4bf(s4[i]);
}

// Async global->LDS (CDNA5, ASYNCcnt-tracked): per-lane 16B copy, no VGPR data.
__device__ __forceinline__ void async_g2l_b128(unsigned lds_off,
                                               const void* gaddr) {
  asm volatile("global_load_async_to_lds_b128 %0, %1, off"
               :: "v"(lds_off), "v"((unsigned long long)gaddr) : "memory");
}
__device__ __forceinline__ void wait_asynccnt0() {
  asm volatile("s_wait_asynccnt 0" ::: "memory");
}

// A-fragment (16x32 bf16, ISA 7.12.2): lane m=lane%16, g=lane/16;
//   halves = row[k0+8g .. +7] ++ row[k0+16+8g .. +7]
// B-fragment (32x16 bf16): lane n=lane%16, g=lane/16; halves = Wrow_n[k0+8g .. +7] ++ Wrow_n[k0+16+8g .. +7]
__device__ __forceinline__ v16bf load_frag2(const unsigned short* p0,
                                            const unsigned short* p1) {
  union { v16bf v; uint4v u[2]; } r;
  r.u[0] = *reinterpret_cast<const uint4v*>(p0);
  r.u[1] = *reinterpret_cast<const uint4v*>(p1);
  return r.v;
}

// One wave: 16-row slab x 128 output cols -> acc[8] 16x16 tiles (f32).
__device__ __forceinline__ void wave_gemm_slab(
    const unsigned short* __restrict__ Aslab, int lda,
    const unsigned short* __restrict__ W, int ldw,
    int K, int lane, v8f acc[8]) {
  const int m = lane & 15;
  const int g = lane >> 4;
  const unsigned short* arow = Aslab + m * lda;
#pragma unroll
  for (int k0 = 0; k0 < 128; k0 += 32) {
    if (k0 >= K) break;
    v16bf af = load_frag2(arow + k0 + 8 * g, arow + k0 + 16 + 8 * g);
#pragma unroll
    for (int j = 0; j < 8; ++j) {
      const unsigned short* wrow = W + (j * 16 + m) * ldw + k0 + 8 * g;
      v16bf bfr = load_frag2(wrow, wrow + 16);
      acc[j] = __builtin_amdgcn_wmma_f32_16x16x32_bf16(
          false, af, false, bfr, (short)0, acc[j], false, false);
    }
  }
}

__global__ __launch_bounds__(THREADS)
void htgnn_temporal_kernel(
    const float* __restrict__ inter, const float* __restrict__ x,
    const float* __restrict__ proj_w, const float* __restrict__ proj_b,
    const float* __restrict__ q_w,   const float* __restrict__ k_w,
    const float* __restrict__ v_w,   const float* __restrict__ fc_w,
    const float* __restrict__ fc_b,  const float* __restrict__ res_w,
    const float* __restrict__ res_b, const float* __restrict__ res_alpha,
    const float* __restrict__ ln_g,  const float* __restrict__ ln_b,
    float* __restrict__ out) {
  __shared__ __align__(16) unsigned short sW[NH * NH];       // streamed weight (bf16)
  __shared__ __align__(16) unsigned short sA[NH * NH];       // inter tile / attn output
  __shared__ __align__(16) unsigned short sH[NH * NH];       // proj output; later x
  __shared__ __align__(16) unsigned short sQK[2 * NH * QLD]; // staging / Q,K / sOut
  __shared__ __align__(16) unsigned short sV[NH * QLD];
  __shared__ float sPE[T_STEPS * NH];
  __shared__ float sProjB[NH], sFcB[NH], sResB[NH], sLnG[NH], sLnB[NH];
  __shared__ float sMu[NH], sRinv[NH];

  unsigned short* sQ = sQK;
  unsigned short* sK = sQK + NH * QLD;
  float* sOut = reinterpret_cast<float*>(sQK);               // 64KB overlay (Q/K dead)

  const int tid  = threadIdx.x;
  const int wave = tid >> 5;
  const int lane = tid & 31;
  const int node0 = blockIdx.x * NODES_PER_WG;
  const int m0 = (lane >> 4) * 8;        // D-layout: vgpr i -> row m0+i
  const int nC = lane & 15;              // D-layout: col = jtile*16 + nC
  const int rbase = wave * 16 + m0;
  const float alpha = 1.f / (1.f + expf(-res_alpha[0]));

  // ---- stage 0: async-copy inter tile; prefetch weights; PE/bias/proj_w ----
  {
    float* staging = reinterpret_cast<float*>(sQK);          // 64KB fp32 staging
    for (int i4 = tid; i4 < (NH * NH) >> 2; i4 += THREADS) {
      int f = i4 << 2;
      int r = f >> 7, c = f & (NH - 1);
      int nl = r >> 3, t = r & 7;
      async_g2l_b128((unsigned)(size_t)(staging + f),
                     inter + ((size_t)t * NNODES + node0 + nl) * NH + c);
    }
    const int line = tid * 64;           // one 256B line per thread covers 128x128
    if (line < NH * NH) {
      __builtin_prefetch(q_w + line, 0, 1);
      __builtin_prefetch(k_w + line, 0, 1);
      __builtin_prefetch(v_w + line, 0, 1);
      __builtin_prefetch(fc_w + line, 0, 1);
    }
    if (line < NH * NI) __builtin_prefetch(res_w + line, 0, 1);

    for (int i = tid; i < T_STEPS * NH; i += THREADS) {
      int t = i >> 7, d = i & (NH - 1);
      float div = expf(-(float)(d & ~1) * (11.512925464970229f / 128.0f));
      float ang = (float)(t + 1) * div;
      sPE[i] = (d & 1) ? cosf(ang) : sinf(ang);
    }
    if (tid < NH) {
      sProjB[tid] = proj_b[tid]; sFcB[tid] = fc_b[tid]; sResB[tid] = res_b[tid];
      sLnG[tid] = ln_g[tid];     sLnB[tid] = ln_b[tid];
    }
    load_w_bf16(sW, proj_w, NH * NH, tid);

    wait_asynccnt0();                    // this wave's async copies landed
    __syncthreads();                     // all waves' copies landed
    const float4* s4 = reinterpret_cast<const float4*>(staging);
    for (int i4 = tid; i4 < (NH * NH) >> 2; i4 += THREADS)
      reinterpret_cast<uint2*>(sA)[i4] = pack4bf(s4[i4]);    // fp32 LDS -> bf16 LDS
  }
  __syncthreads();

  // ---- proj: H = A @ Wproj^T + b + pe ----
  {
    v8f acc[8];
#pragma unroll
    for (int j = 0; j < 8; ++j) { v8f z = {0,0,0,0,0,0,0,0}; acc[j] = z; }
    wave_gemm_slab(sA + wave * 16 * NH, NH, sW, NH, NH, lane, acc);
#pragma unroll
    for (int j = 0; j < 8; ++j)
#pragma unroll
      for (int i2 = 0; i2 < 8; ++i2) {
        int r = rbase + i2, col = j * 16 + nC;
        sH[r * NH + col] = f2bf_us(acc[j][i2] + sProjB[col] + sPE[(r & 7) * NH + col]);
      }
  }
  __syncthreads();

  // ---- q, k, v projections (weight buffer streamed per stage) ----
  const float* wptrs[3] = {q_w, k_w, v_w};
  unsigned short* dsts[3] = {sQ, sK, sV};
  for (int s = 0; s < 3; ++s) {
    load_w_bf16(sW, wptrs[s], NH * NH, tid);
    __syncthreads();
    v8f acc[8];
#pragma unroll
    for (int j = 0; j < 8; ++j) { v8f z = {0,0,0,0,0,0,0,0}; acc[j] = z; }
    wave_gemm_slab(sH + wave * 16 * NH, NH, sW, NH, NH, lane, acc);
    unsigned short* dst = dsts[s];
#pragma unroll
    for (int j = 0; j < 8; ++j)
#pragma unroll
      for (int i2 = 0; i2 < 8; ++i2)
        dst[(rbase + i2) * QLD + j * 16 + nC] = f2bf_us(acc[j][i2]);
    __syncthreads();
  }

  // ---- attention (8x8 per node, VALU) + stage fc_w and x ----
  load_w_bf16(sW, fc_w, NH * NH, tid);
  for (int i4 = tid; i4 < (NH * NI) >> 2; i4 += THREADS) {   // x -> sH (stride NI)
    int f = i4 << 2;
    int r = f >> 6, c = f & (NI - 1);
    int nl = r >> 3, t = r & 7;
    float4 v = *reinterpret_cast<const float4*>(
        x + ((size_t)t * NNODES + node0 + nl) * NI + c);
    reinterpret_cast<uint2*>(sH)[i4] = pack4bf(v);
  }
  if (tid < 128) {
    const int r = tid, base = r & ~7;
    float sc[8]; float mx = -1e30f;
    const unsigned short* qp = sQ + r * QLD;
#pragma unroll
    for (int j = 0; j < 8; ++j) {
      const unsigned short* kp = sK + (base + j) * QLD;
      float s = 0.f;
      for (int c = 0; c < NH; ++c) s += bf2f(qp[c]) * bf2f(kp[c]);
      sc[j] = s; mx = fmaxf(mx, s);
    }
    float den = 0.f;
#pragma unroll
    for (int j = 0; j < 8; ++j) { sc[j] = expf(sc[j] - mx); den += sc[j]; }
    float inv = 1.f / den;
    for (int c = 0; c < NH; ++c) {
      float o = 0.f;
#pragma unroll
      for (int j = 0; j < 8; ++j) o += sc[j] * bf2f(sV[(base + j) * QLD + c]);
      sA[r * NH + c] = f2bf_us(o * inv);
    }
  }
  __syncthreads();

  // ---- fc: h_ = relu(attn @ Wfc^T + b) -> sOut (overlays dead Q/K) ----
  {
    v8f acc[8];
#pragma unroll
    for (int j = 0; j < 8; ++j) { v8f z = {0,0,0,0,0,0,0,0}; acc[j] = z; }
    wave_gemm_slab(sA + wave * 16 * NH, NH, sW, NH, NH, lane, acc);
#pragma unroll
    for (int j = 0; j < 8; ++j)
#pragma unroll
      for (int i2 = 0; i2 < 8; ++i2) {
        int r = rbase + i2, col = j * 16 + nC;
        sOut[r * NH + col] = fmaxf(acc[j][i2] + sFcB[col], 0.f);
      }
  }
  __syncthreads();

  // ---- residual: res = x @ Wres^T + b ; out = h_*a + res*(1-a) ----
  load_w_bf16(sW, res_w, NH * NI, tid);
  __syncthreads();
  {
    v8f acc[8];
#pragma unroll
    for (int j = 0; j < 8; ++j) { v8f z = {0,0,0,0,0,0,0,0}; acc[j] = z; }
    wave_gemm_slab(sH + wave * 16 * NI, NI, sW, NI, NI, lane, acc);
    const float oma = 1.f - alpha;
#pragma unroll
    for (int j = 0; j < 8; ++j)
#pragma unroll
      for (int i2 = 0; i2 < 8; ++i2) {
        int r = rbase + i2, col = j * 16 + nC;
        float resv = acc[j][i2] + sResB[col];
        sOut[r * NH + col] = sOut[r * NH + col] * alpha + resv * oma;
      }
  }
  __syncthreads();

  // ---- LayerNorm: per-row stats, then coalesced float4 stores ----
  if (tid < 128) {
    const int r = tid;
    const float* row = sOut + r * NH;
    float mu = 0.f;
    for (int c = 0; c < NH; ++c) mu += row[c];
    mu *= (1.f / NH);
    float var = 0.f;
    for (int c = 0; c < NH; ++c) { float d = row[c] - mu; var += d * d; }
    var *= (1.f / NH);
    sMu[r] = mu;
    sRinv[r] = rsqrtf(var + 1e-5f);
  }
  __syncthreads();
  for (int i4 = tid; i4 < (NH * NH) >> 2; i4 += THREADS) {
    int f = i4 << 2;
    int r = f >> 7, c = f & (NH - 1);
    float mu = sMu[r], rinv = sRinv[r];
    float4 v = *reinterpret_cast<const float4*>(sOut + f);
    v.x = (v.x - mu) * rinv * sLnG[c + 0] + sLnB[c + 0];
    v.y = (v.y - mu) * rinv * sLnG[c + 1] + sLnB[c + 1];
    v.z = (v.z - mu) * rinv * sLnG[c + 2] + sLnB[c + 2];
    v.w = (v.w - mu) * rinv * sLnG[c + 3] + sLnB[c + 3];
    int nl = r >> 3, t = r & 7;
    *reinterpret_cast<float4*>(out + ((size_t)t * NNODES + node0 + nl) * NH + c) = v;
  }
}

extern "C" void kernel_launch(void* const* d_in, const int* in_sizes, int n_in,
                              void* d_out, int out_size, void* d_ws, size_t ws_size,
                              hipStream_t stream) {
  const float* inter     = (const float*)d_in[0];
  const float* x         = (const float*)d_in[1];
  const float* proj_w    = (const float*)d_in[2];
  const float* proj_b    = (const float*)d_in[3];
  const float* q_w       = (const float*)d_in[4];
  const float* k_w       = (const float*)d_in[5];
  const float* v_w       = (const float*)d_in[6];
  const float* fc_w      = (const float*)d_in[7];
  const float* fc_b      = (const float*)d_in[8];
  const float* res_w     = (const float*)d_in[9];
  const float* res_b     = (const float*)d_in[10];
  const float* res_alpha = (const float*)d_in[11];
  const float* ln_g      = (const float*)d_in[12];
  const float* ln_b      = (const float*)d_in[13];
  dim3 grid(NNODES / NODES_PER_WG);   // 6250 workgroups, 16 nodes each
  htgnn_temporal_kernel<<<grid, THREADS, 0, stream>>>(
      inter, x, proj_w, proj_b, q_w, k_w, v_w, fc_w, fc_b,
      res_w, res_b, res_alpha, ln_g, ln_b, (float*)d_out);
}
